// MultiHeadAttention_53807350284415
// MI455X (gfx1250) — compile-verified, matrix operand order checked by reading, not run on hardware
//
#include <hip/hip_runtime.h>
#include <hip/hip_bf16.h>
#include <math.h>

// MHA forward for B=2,S=2048,E=1024,H=16,D=64 on gfx1250 (wave32, WMMA).
// All GEMMs via v_wmma_f32_16x16x32_bf16 (bf16 in, f32 accumulate).

typedef __attribute__((ext_vector_type(16))) __bf16 v16bf;
typedef __attribute__((ext_vector_type(8)))  float  v8f;

#define B_  2
#define S_  2048
#define E_  1024
#define H_  16
#define D_  64
#define HD_ 1024
#define N_  (B_*S_)   // 4096 tokens

static __device__ __forceinline__ v8f wmma_bf16(v16bf a, v16bf b, v8f c) {
  // D = A(16x32) * B(32x16) + C(16x16 f32)
  return __builtin_amdgcn_wmma_f32_16x16x32_bf16(false, a, false, b,
                                                 (short)0, c, false, false);
}

// ---------------- conversion kernels ----------------
__global__ void cvt_f32_bf16(const float* __restrict__ src,
                             __bf16* __restrict__ dst, int n) {
  int i = blockIdx.x * blockDim.x + threadIdx.x;
  int stride = gridDim.x * blockDim.x;
  for (; i < n; i += stride) dst[i] = (__bf16)src[i];
}

// src [rows, cols] row-major f32 -> dst [cols, rows] bf16 (transpose)
__global__ void cvt_transpose_bf16(const float* __restrict__ src,
                                   __bf16* __restrict__ dst,
                                   int rows, int cols) {
  int i = blockIdx.x * blockDim.x + threadIdx.x;
  int stride = gridDim.x * blockDim.x;
  int n = rows * cols;
  for (; i < n; i += stride) {
    int r = i / cols, c = i % cols;
    dst[(size_t)c * rows + r] = (__bf16)src[i];
  }
}

// ---------------- QKV projection + RoPE ----------------
// One wave computes a 16-token x 64 (full head) tile of one of q/k/v.
// wt: transposed weight [HD][E] bf16. which: 0=q (RoPE+scale), 1=k (RoPE), 2=v (transposed store).
__global__ void qkv_rope_kernel(const __bf16* __restrict__ xb,
                                const __bf16* __restrict__ wt,
                                const float*  __restrict__ bias,
                                __bf16* __restrict__ dst,
                                int which) {
  const int wave = threadIdx.x >> 5;
  const int lane = threadIdx.x & 31;
  const int l15  = lane & 15;
  const int hi   = lane >> 4;
  const int rowtile = blockIdx.x * 4 + wave;
  const int n0 = rowtile * 16;
  const int h  = blockIdx.y;
  const int b  = n0 / S_;
  const int s0 = n0 % S_;          // tiles never straddle batches (S % 16 == 0)
  const int bh = b * H_ + h;

  v8f acc0 = {}, acc1 = {}, acc2 = {}, acc3 = {};
  const __bf16* aptr  = xb + (size_t)(n0 + l15) * E_ + hi * 16;
  const __bf16* bptr0 = wt + (size_t)(h * 64 +  0 + l15) * E_ + hi * 16;
  const __bf16* bptr1 = wt + (size_t)(h * 64 + 16 + l15) * E_ + hi * 16;
  const __bf16* bptr2 = wt + (size_t)(h * 64 + 32 + l15) * E_ + hi * 16;
  const __bf16* bptr3 = wt + (size_t)(h * 64 + 48 + l15) * E_ + hi * 16;

  for (int kk = 0; kk < E_ / 32; ++kk) {
    __builtin_prefetch(aptr + 64, 0, 0);   // global_prefetch_b8 next tiles
    v16bf a  = *(const v16bf*)aptr;
    v16bf b0 = *(const v16bf*)bptr0;
    v16bf b1 = *(const v16bf*)bptr1;
    v16bf b2 = *(const v16bf*)bptr2;
    v16bf b3 = *(const v16bf*)bptr3;
    acc0 = wmma_bf16(a, b0, acc0);
    acc1 = wmma_bf16(a, b1, acc1);
    acc2 = wmma_bf16(a, b2, acc2);
    acc3 = wmma_bf16(a, b3, acc3);
    aptr += 32; bptr0 += 32; bptr1 += 32; bptr2 += 32; bptr3 += 32;
  }

  // bias (per output column)
  {
    float b0 = bias[h * 64 +  0 + l15];
    float b1 = bias[h * 64 + 16 + l15];
    float b2 = bias[h * 64 + 32 + l15];
    float b3 = bias[h * 64 + 48 + l15];
#pragma unroll
    for (int r = 0; r < 8; ++r) {
      acc0[r] += b0; acc1[r] += b1; acc2[r] += b2; acc3[r] += b3;
    }
  }

  if (which < 2) {
    // RoPE: pairs (d, d+32): chunk0<->chunk2, chunk1<->chunk3.
#pragma unroll
    for (int c = 0; c < 2; ++c) {
      int d0 = c * 16 + l15;                         // 0..31
      float ts = __powf(10000.0f, -(float)d0 * (1.0f / 32.0f));
      v8f& lo = (c == 0) ? acc0 : acc1;
      v8f& up = (c == 0) ? acc2 : acc3;
#pragma unroll
      for (int r = 0; r < 8; ++r) {
        float spos = (float)(s0 + r + 8 * hi);
        float th = spos * ts;
        float cs = __cosf(th), sn = __sinf(th);
        float v0 = lo[r], v2 = up[r];
        lo[r] = v0 * cs - v2 * sn;
        up[r] = v2 * cs + v0 * sn;
      }
    }
    if (which == 0) {                    // fold 1/sqrt(D) into q
#pragma unroll
      for (int r = 0; r < 8; ++r) {
        acc0[r] *= 0.125f; acc1[r] *= 0.125f; acc2[r] *= 0.125f; acc3[r] *= 0.125f;
      }
    }
  }

#pragma unroll
  for (int c = 0; c < 4; ++c) {
    const v8f acc = (c == 0) ? acc0 : (c == 1) ? acc1 : (c == 2) ? acc2 : acc3;
    int d = c * 16 + l15;
#pragma unroll
    for (int r = 0; r < 8; ++r) {
      int s = s0 + r + 8 * hi;
      size_t idx = (which == 2)
                     ? ((size_t)bh * D_ + d) * S_ + s   // V stored transposed [B,H,D,S]
                     : ((size_t)bh * S_ + s) * D_ + d;  // Q/K stored [B,H,S,D]
      dst[idx] = (__bf16)acc[r];
    }
  }
}

// ---------------- flash attention (causal + pad mask) ----------------
__global__ void attn_kernel(const __bf16* __restrict__ qb,
                            const __bf16* __restrict__ kb,
                            const __bf16* __restrict__ vt,
                            const int*    __restrict__ mask,
                            __bf16* __restrict__ ctxb) {
  __shared__ __attribute__((aligned(32))) __bf16 plds[4][16][32];
  const int wave = threadIdx.x >> 5;
  const int lane = threadIdx.x & 31;
  const int l15  = lane & 15;
  const int hi   = lane >> 4;
  const int qtile = blockIdx.x * 4 + wave;
  const int h = blockIdx.y;
  const int b = blockIdx.z;
  const int bh = b * H_ + h;
  const int q0 = qtile * 16;

  // Q fragments (A layout): lane row = q0 + l15, k-chunk = hi*16
  const __bf16* qrow = qb + ((size_t)bh * S_ + q0 + l15) * D_;
  v16bf qa0 = *(const v16bf*)(qrow + hi * 16);        // d = 0..31
  v16bf qa1 = *(const v16bf*)(qrow + 32 + hi * 16);   // d = 32..63

  v8f acc0 = {}, acc1 = {}, acc2 = {}, acc3 = {};
  float mrun[8], lrun[8];
#pragma unroll
  for (int r = 0; r < 8; ++r) { mrun[r] = -3.0e38f; lrun[r] = 0.0f; }

  const int ktmax = (q0 + 15) / 32;
  for (int kt = 0; kt <= ktmax; ++kt) {
    const int cb = kt * 32;
    // K fragments (B layout): lane col = key, k = d
    const __bf16* kr0 = kb + ((size_t)bh * S_ + cb + l15) * D_ + hi * 16;
    const __bf16* kr1 = kr0 + 16 * D_;
    v16bf kb00 = *(const v16bf*)(kr0);
    v16bf kb01 = *(const v16bf*)(kr0 + 32);
    v16bf kb10 = *(const v16bf*)(kr1);
    v16bf kb11 = *(const v16bf*)(kr1 + 32);

    v8f s0v = {}, s1v = {};
    s0v = wmma_bf16(qa0, kb00, s0v);
    s0v = wmma_bf16(qa1, kb01, s0v);
    s1v = wmma_bf16(qa0, kb10, s1v);
    s1v = wmma_bf16(qa1, kb11, s1v);

    const int col0 = cb + l15, col1 = col0 + 16;
    const bool mk0 = mask[b * S_ + col0] != 0;
    const bool mk1 = mask[b * S_ + col1] != 0;
    const float NEG = -3.0e38f;
    float p0[8], p1[8];
#pragma unroll
    for (int r = 0; r < 8; ++r) {
      int row = q0 + r + 8 * hi;
      float v0 = (mk0 && col0 <= row) ? s0v[r] : NEG;
      float v1 = (mk1 && col1 <= row) ? s1v[r] : NEG;
      float tm = fmaxf(v0, v1);
      tm = fmaxf(tm, __shfl_xor(tm, 1, 32));
      tm = fmaxf(tm, __shfl_xor(tm, 2, 32));
      tm = fmaxf(tm, __shfl_xor(tm, 4, 32));
      tm = fmaxf(tm, __shfl_xor(tm, 8, 32));
      float mn = fmaxf(mrun[r], tm);
      float alpha = __expf(mrun[r] - mn);
      float e0 = __expf(v0 - mn);
      float e1 = __expf(v1 - mn);
      float ps = e0 + e1;
      ps += __shfl_xor(ps, 1, 32);
      ps += __shfl_xor(ps, 2, 32);
      ps += __shfl_xor(ps, 4, 32);
      ps += __shfl_xor(ps, 8, 32);
      lrun[r] = lrun[r] * alpha + ps;
      mrun[r] = mn;
      acc0[r] *= alpha; acc1[r] *= alpha; acc2[r] *= alpha; acc3[r] *= alpha;
      p0[r] = e0; p1[r] = e1;
    }

    // Re-layout P (C layout -> A layout) through LDS
#pragma unroll
    for (int r = 0; r < 8; ++r) {
      plds[wave][8 * hi + r][l15]      = (__bf16)p0[r];
      plds[wave][8 * hi + r][l15 + 16] = (__bf16)p1[r];
    }
    asm volatile("s_wait_dscnt 0" ::: "memory");
    v16bf pa = *(const v16bf*)&plds[wave][l15][hi * 16];

    // V fragments (B layout) from transposed V: lane col = d, k = key
    const __bf16* vr = vt + ((size_t)bh * D_ + l15) * S_ + cb + hi * 16;
    v16bf vb0 = *(const v16bf*)(vr);
    v16bf vb1 = *(const v16bf*)(vr + 16 * S_);
    v16bf vb2 = *(const v16bf*)(vr + 32 * S_);
    v16bf vb3 = *(const v16bf*)(vr + 48 * S_);
    acc0 = wmma_bf16(pa, vb0, acc0);
    acc1 = wmma_bf16(pa, vb1, acc1);
    acc2 = wmma_bf16(pa, vb2, acc2);
    acc3 = wmma_bf16(pa, vb3, acc3);
  }

  // normalize and store ctx as bf16 [N, HD]
#pragma unroll
  for (int r = 0; r < 8; ++r) {
    float inv = 1.0f / lrun[r];
    int n = b * S_ + q0 + r + 8 * hi;
    size_t base = (size_t)n * HD_ + h * 64 + l15;
    ctxb[base +  0] = (__bf16)(acc0[r] * inv);
    ctxb[base + 16] = (__bf16)(acc1[r] * inv);
    ctxb[base + 32] = (__bf16)(acc2[r] * inv);
    ctxb[base + 48] = (__bf16)(acc3[r] * inv);
  }
}

// ---------------- output projection ----------------
__global__ void out_proj_kernel(const __bf16* __restrict__ ctxb,
                                const __bf16* __restrict__ wot,  // [E][HD] transposed
                                const float*  __restrict__ bo,
                                float* __restrict__ out) {
  const int wave = threadIdx.x >> 5;
  const int lane = threadIdx.x & 31;
  const int l15  = lane & 15;
  const int hi   = lane >> 4;
  const int n0 = blockIdx.x * 16;
  const int colchunk = blockIdx.y * 4 + wave;
  const int c0 = colchunk * 16;
  const __bf16* aptr = ctxb + (size_t)(n0 + l15) * HD_ + hi * 16;
  const __bf16* bptr = wot  + (size_t)(c0 + l15) * HD_ + hi * 16;
  v8f acc = {};
  for (int kk = 0; kk < HD_ / 32; ++kk) {
    __builtin_prefetch(bptr + 64, 0, 0);
    v16bf a = *(const v16bf*)aptr;
    v16bf bfrag = *(const v16bf*)bptr;
    acc = wmma_bf16(a, bfrag, acc);
    aptr += 32; bptr += 32;
  }
  float bias = bo[c0 + l15];
#pragma unroll
  for (int r = 0; r < 8; ++r) {
    out[(size_t)(n0 + r + 8 * hi) * E_ + c0 + l15] = acc[r] + bias;
  }
}

// ---------------- launcher ----------------
extern "C" void kernel_launch(void* const* d_in, const int* in_sizes, int n_in,
                              void* d_out, int out_size, void* d_ws, size_t ws_size,
                              hipStream_t stream) {
  const float* x    = (const float*)d_in[0];
  const int*   mask = (const int*)  d_in[1];
  const float* Wq   = (const float*)d_in[2];
  const float* bq   = (const float*)d_in[3];
  const float* Wk   = (const float*)d_in[4];
  const float* bk   = (const float*)d_in[5];
  const float* Wv   = (const float*)d_in[6];
  const float* bv   = (const float*)d_in[7];
  const float* Wo   = (const float*)d_in[8];
  const float* bo   = (const float*)d_in[9];
  float* out = (float*)d_out;

  char* ws = (char*)d_ws;
  // workspace layout (bytes): total 48 MiB
  __bf16* xb  = (__bf16*)(ws);                    // 8 MiB  x bf16 [N,E]
  __bf16* wqt = (__bf16*)(ws + 8388608);          // 2 MiB  Wq^T
  __bf16* wkt = (__bf16*)(ws + 10485760);         // 2 MiB  Wk^T
  __bf16* wvt = (__bf16*)(ws + 12582912);         // 2 MiB  Wv^T
  __bf16* wot = (__bf16*)(ws + 14680064);         // 2 MiB  Wo^T
  __bf16* qb  = (__bf16*)(ws + 16777216);         // 8 MiB  q [B,H,S,D]
  __bf16* kb  = (__bf16*)(ws + 25165824);         // 8 MiB  k [B,H,S,D]
  __bf16* vt  = (__bf16*)(ws + 33554432);         // 8 MiB  v [B,H,D,S]
  __bf16* ctx = (__bf16*)(ws + 41943040);         // 8 MiB  ctx [N,HD]

  cvt_f32_bf16<<<2048, 256, 0, stream>>>(x, xb, N_ * E_);
  cvt_transpose_bf16<<<1024, 256, 0, stream>>>(Wq, wqt, E_, HD_);
  cvt_transpose_bf16<<<1024, 256, 0, stream>>>(Wk, wkt, E_, HD_);
  cvt_transpose_bf16<<<1024, 256, 0, stream>>>(Wv, wvt, E_, HD_);
  cvt_transpose_bf16<<<1024, 256, 0, stream>>>(Wo, wot, HD_, E_);

  dim3 gq(N_ / 16 / 4, H_);            // (64, 16), 4 waves/block
  qkv_rope_kernel<<<gq, 128, 0, stream>>>(xb, wqt, bq, qb, 0);
  qkv_rope_kernel<<<gq, 128, 0, stream>>>(xb, wkt, bk, kb, 1);
  qkv_rope_kernel<<<gq, 128, 0, stream>>>(xb, wvt, bv, vt, 2);

  dim3 ga(S_ / 16 / 4, H_, B_);        // (32, 16, 2)
  attn_kernel<<<ga, 128, 0, stream>>>(qb, kb, vt, mask, ctx);

  dim3 go(N_ / 16, HD_ / 16 / 4);      // (256, 16)
  out_proj_kernel<<<go, 128, 0, stream>>>(ctx, wot, bo, out);
}
